// NGMConvLayer_76819785056587
// MI455X (gfx1250) — matrix operand order repaired and checked by reading, not compiled, not run on hardware
//
#include <hip/hip_runtime.h>
#include <hip/hip_bf16.h>

#define N_NODES   65536
#define IN_C      512
#define OUT_C     256
#define E_EDGES   65536
#define E_TOT     131072   // edges + self loops
#define NEG_SLOPE 0.2f

typedef __attribute__((ext_vector_type(16))) __bf16 v16bf;
typedef __attribute__((ext_vector_type(8)))  float  v8f;

union AB32 { v16bf v; uint4 q[2]; };

// ---- order-preserving float <-> uint key (for atomic segment max) ----
__device__ __forceinline__ unsigned fkey(float f) {
    unsigned u = __float_as_uint(f);
    return (u & 0x80000000u) ? ~u : (u | 0x80000000u);
}
__device__ __forceinline__ float keyf(unsigned k) {
    unsigned u = (k & 0x80000000u) ? (k & 0x7FFFFFFFu) : ~k;
    return __uint_as_float(u);
}

// ============================================================
// 1) convert x (f32, row-major N x 512) -> bf16
// ============================================================
__global__ void cvt_x_kernel(const float* __restrict__ x, __bf16* __restrict__ xb, int total) {
    int i = (blockIdx.x * blockDim.x + threadIdx.x) * 4;
    if (i < total) {
        float4 f = *(const float4*)(x + i);
        xb[i + 0] = (__bf16)f.x;
        xb[i + 1] = (__bf16)f.y;
        xb[i + 2] = (__bf16)f.z;
        xb[i + 3] = (__bf16)f.w;
    }
}

// ============================================================
// 2) pack W_l / W_r / W_self (512x256 f32, K-major) into the
//    CDNA5 bf16 B-operand lane layout, 1KB per (kb,nb) tile:
//    lane n' (0..15) = col nb*16+n', K = {0..7, 16..23} of kb*32
//    lane n'+16      = same col,     K = {8..15, 24..31}
//    16 contiguous bf16 (32B) per lane.
// ============================================================
__global__ void pack_w_kernel(const float* __restrict__ Wl,
                              const float* __restrict__ Wr,
                              const float* __restrict__ Ws,
                              __bf16* __restrict__ wpack) {
    int tid = blockIdx.x * blockDim.x + threadIdx.x;
    if (tid >= 3 * 256 * 32) return;
    int lane = tid & 31;
    int blk  = (tid >> 5) & 255;          // kb*16 + nb
    int mat  = tid >> 13;                 // 0=l,1=r,2=self
    int kb = blk >> 4, nb = blk & 15;
    const float* W = (mat == 0) ? Wl : ((mat == 1) ? Wr : Ws);
    int n  = nb * 16 + (lane & 15);
    int k0 = kb * 32 + (lane >> 4) * 8;
    __bf16* outp = wpack + ((size_t)(mat * 256 + blk) * 32 + lane) * 16;
    #pragma unroll
    for (int i = 0; i < 16; ++i)
        outp[i] = (__bf16)W[(size_t)(k0 + i + (i & 8)) * OUT_C + n];
}

// ============================================================
// 3) zero-init segment-max keys and denominators
// ============================================================
__global__ void zero_kernel(unsigned* __restrict__ mkey, float* __restrict__ denom) {
    int i = blockIdx.x * blockDim.x + threadIdx.x;
    if (i < N_NODES) { mkey[i] = 0u; denom[i] = 0.0f; }
}

// ============================================================
// 4) fused triple GEMM:  x_l = x@W_l+b_l ; x_r = x@W_r+b_r ;
//    x_out(base) = x@W_self + b_self + conv_bias
//    Block = 8 waves, each wave owns one 16-row M tile.
//    B panel (12 tiles = 12KB per k-step, shared by all 8 waves)
//    double-buffered in LDS with COMPILE-TIME buffer indices
//    (two-phase pipeline) so accumulators never move registers.
// ============================================================
__global__ void __launch_bounds__(256) gemm3_kernel(
                             const __bf16* __restrict__ xb,
                             const __bf16* __restrict__ wpack,
                             const float* __restrict__ b_l,
                             const float* __restrict__ b_r,
                             const float* __restrict__ b_self,
                             const float* __restrict__ conv_bias,
                             float* __restrict__ xl,
                             float* __restrict__ xr,
                             float* __restrict__ xout) {
    __shared__ uint4 sB[2][768];          // 2 x 12KB double buffer

    const int tid    = threadIdx.x;
    const int lane   = tid & 31;
    const int wid    = tid >> 5;
    const int ngroup = blockIdx.x & 3;    // 4 groups of 64 output cols
    const int mtile  = (blockIdx.x >> 2) * 8 + wid;

    // --- staging: thread copies chunks c = tid + r*256 of the 768-chunk panel.
    // chunk c -> tile ti = c/64 (= mat*4 + t), 16B piece w = c%64.
    // global tile (mat, kb, nb=ngroup*4+t) at byte (mat*256+kb*16+nb)*1024;
    // kb stride = 16384 bytes.
    const char* gsrc[3];
    #pragma unroll
    for (int r = 0; r < 3; ++r) {
        int c   = tid + r * 256;
        int ti  = c >> 6;
        int w   = c & 63;
        int mat = ti >> 2, t = ti & 3;
        gsrc[r] = (const char*)wpack
                + (size_t)(mat * 256 + ngroup * 4 + t) * 1024 + (size_t)w * 16;
    }

    // stage k-step 0 into sB[0]
    #pragma unroll
    for (int r = 0; r < 3; ++r)
        sB[0][tid + r * 256] = *(const uint4*)gsrc[r];

    // A-operand addressing per ISA 16-bit A layout:
    // lanes 0-15 : row = m+lane,    K chunks {0..7, 16..23}
    // lanes 16-31: row = m+lane-16, K chunks {8..15, 24..31}
    const int mrow = mtile * 16 + (lane & 15);
    const int kSel = (lane >> 4) * 8;
    const __bf16* ap = xb + (size_t)mrow * IN_C + kSel;

    AB32 a0, a1;
    a0.q[0] = *(const uint4*)(ap);
    a0.q[1] = *(const uint4*)(ap + 16);

    v8f zero = {};
    v8f acc[12];                           // [0..3]=l, [4..7]=r, [8..11]=self
    #pragma unroll
    for (int i = 0; i < 12; ++i) acc[i] = zero;

    __syncthreads();

    for (int ks = 0; ks < 8; ++ks) {
        // ---- phase 0: compute k-step 2*ks from sB[0];
        //      prefetch k-step 2*ks+1 into sB[1] and a1.
        {
            const __bf16* apx = ap + (size_t)(2 * ks + 1) * 32;
            a1.q[0] = *(const uint4*)(apx);
            a1.q[1] = *(const uint4*)(apx + 16);
            #pragma unroll
            for (int r = 0; r < 3; ++r)
                sB[1][tid + r * 256] =
                    *(const uint4*)(gsrc[r] + (size_t)(2 * ks + 1) * 16384);
            #pragma unroll
            for (int ti = 0; ti < 12; ++ti) {
                AB32 b;
                b.q[0] = sB[0][ti * 64 + lane * 2];
                b.q[1] = sB[0][ti * 64 + lane * 2 + 1];
                acc[ti] = __builtin_amdgcn_wmma_f32_16x16x32_bf16(
                              false, a0.v, false, b.v, (short)0, acc[ti], false, false);
            }
            __syncthreads();
        }
        // ---- phase 1: compute k-step 2*ks+1 from sB[1];
        //      prefetch k-step 2*ks+2 into sB[0] and a0 (except last).
        {
            if (ks < 7) {
                const __bf16* apx = ap + (size_t)(2 * ks + 2) * 32;
                a0.q[0] = *(const uint4*)(apx);
                a0.q[1] = *(const uint4*)(apx + 16);
                #pragma unroll
                for (int r = 0; r < 3; ++r)
                    sB[0][tid + r * 256] =
                        *(const uint4*)(gsrc[r] + (size_t)(2 * ks + 2) * 16384);
            }
            #pragma unroll
            for (int ti = 0; ti < 12; ++ti) {
                AB32 b;
                b.q[0] = sB[1][ti * 64 + lane * 2];
                b.q[1] = sB[1][ti * 64 + lane * 2 + 1];
                acc[ti] = __builtin_amdgcn_wmma_f32_16x16x32_bf16(
                              false, a1.v, false, b.v, (short)0, acc[ti], false, false);
            }
            __syncthreads();
        }
    }

    // C/D layout: lanes 0-15 -> rows j (VGPR j), lanes 16-31 -> rows j+8; col = lane%16
    const int nlo   = lane & 15;
    const int rbase = mtile * 16 + ((lane >> 4) ? 8 : 0);
    #pragma unroll
    for (int t = 0; t < 4; ++t) {
        int col = ngroup * 64 + t * 16 + nlo;
        float vbl = b_l[col];
        float vbr = b_r[col];
        float vbs = b_self[col] + conv_bias[col];
        #pragma unroll
        for (int j = 0; j < 8; ++j) {
            size_t idx = (size_t)(rbase + j) * OUT_C + col;
            xl[idx]   = acc[t][j]     + vbl;
            xr[idx]   = acc[4 + t][j] + vbr;
            xout[idx] = acc[8 + t][j] + vbs;
        }
    }
}

// ============================================================
// 5) per-edge attention logit + atomic segment max (wave/edge)
// ============================================================
__global__ void edge_logit_kernel(const float* __restrict__ xl,
                                  const float* __restrict__ xr,
                                  const float* __restrict__ att,
                                  const int* __restrict__ ei,
                                  float* __restrict__ elog,
                                  unsigned* __restrict__ mkey) {
    int wave = (blockIdx.x * blockDim.x + threadIdx.x) >> 5;
    int lane = threadIdx.x & 31;
    if (wave >= E_TOT) return;
    int src, dst;
    if (wave < E_EDGES) { src = ei[wave]; dst = ei[E_EDGES + wave]; }
    else                { src = dst = wave - E_EDGES; }

    const float4* pl = (const float4*)(xl + (size_t)src * OUT_C) + lane * 2;
    const float4* pr = (const float4*)(xr + (size_t)dst * OUT_C) + lane * 2;
    const float4* pa = (const float4*)att + lane * 2;

    float acc = 0.0f;
    #pragma unroll
    for (int h = 0; h < 2; ++h) {
        float4 a = pl[h], b = pr[h], w = pa[h];
        float sx = a.x + b.x, sy = a.y + b.y, sz = a.z + b.z, sw = a.w + b.w;
        sx = sx > 0.0f ? sx : NEG_SLOPE * sx;
        sy = sy > 0.0f ? sy : NEG_SLOPE * sy;
        sz = sz > 0.0f ? sz : NEG_SLOPE * sz;
        sw = sw > 0.0f ? sw : NEG_SLOPE * sw;
        acc += sx * w.x + sy * w.y + sz * w.z + sw * w.w;
    }
    #pragma unroll
    for (int off = 16; off > 0; off >>= 1) acc += __shfl_xor(acc, off, 32);
    if (lane == 0) {
        elog[wave] = acc;
        atomicMax(&mkey[dst], fkey(acc));
    }
}

// ============================================================
// 6) denominators of the segment softmax (thread/edge)
// ============================================================
__global__ void edge_denom_kernel(const float* __restrict__ elog,
                                  const unsigned* __restrict__ mkey,
                                  float* __restrict__ denom,
                                  const int* __restrict__ ei) {
    int i = blockIdx.x * blockDim.x + threadIdx.x;
    if (i >= E_TOT) return;
    int dst = (i < E_EDGES) ? ei[E_EDGES + i] : (i - E_EDGES);
    float p = __expf(elog[i] - keyf(mkey[dst]));
    unsafeAtomicAdd(&denom[dst], p);
}

// ============================================================
// 7) weighted aggregation: xout[dst] += alpha * x_l[src]  (wave/edge)
// ============================================================
__global__ void edge_agg_kernel(const float* __restrict__ elog,
                                const unsigned* __restrict__ mkey,
                                const float* __restrict__ denom,
                                const float* __restrict__ xl,
                                const int* __restrict__ ei,
                                float* __restrict__ xout) {
    int wave = (blockIdx.x * blockDim.x + threadIdx.x) >> 5;
    int lane = threadIdx.x & 31;
    if (wave >= E_TOT) return;
    int src, dst;
    if (wave < E_EDGES) { src = ei[wave]; dst = ei[E_EDGES + wave]; }
    else                { src = dst = wave - E_EDGES; }

    float alpha = __expf(elog[wave] - keyf(mkey[dst])) / denom[dst];
    const float4* pl = (const float4*)(xl + (size_t)src * OUT_C) + lane * 2;
    float* po = xout + (size_t)dst * OUT_C + lane * 8;
    float4 a = pl[0], b = pl[1];
    unsafeAtomicAdd(po + 0, alpha * a.x);
    unsafeAtomicAdd(po + 1, alpha * a.y);
    unsafeAtomicAdd(po + 2, alpha * a.z);
    unsafeAtomicAdd(po + 3, alpha * a.w);
    unsafeAtomicAdd(po + 4, alpha * b.x);
    unsafeAtomicAdd(po + 5, alpha * b.y);
    unsafeAtomicAdd(po + 6, alpha * b.z);
    unsafeAtomicAdd(po + 7, alpha * b.w);
}

// ============================================================
// 8) classifier: s[i] = xout[i,:] . W_cls + b_cls   (wave/node)
// ============================================================
__global__ void cls_dot_kernel(const float* __restrict__ xout,
                               const float* __restrict__ wcls,
                               const float* __restrict__ bcls,
                               float* __restrict__ s) {
    int wave = (blockIdx.x * blockDim.x + threadIdx.x) >> 5;
    int lane = threadIdx.x & 31;
    if (wave >= N_NODES) return;
    const float4* px = (const float4*)(xout + (size_t)wave * OUT_C) + lane * 2;
    const float4* pw = (const float4*)wcls + lane * 2;
    float4 a = px[0], b = px[1], c = pw[0], d = pw[1];
    float acc = a.x * c.x + a.y * c.y + a.z * c.z + a.w * c.w
              + b.x * d.x + b.y * d.y + b.z * d.z + b.w * d.w;
    #pragma unroll
    for (int off = 16; off > 0; off >>= 1) acc += __shfl_xor(acc, off, 32);
    if (lane == 0) s[wave] = acc + bcls[0];
}

// ============================================================
// 9) row softmax over 256x256 grid + broadcast add to xout
// ============================================================
__global__ void sk_softmax_kernel(const float* __restrict__ s,
                                  float* __restrict__ xout) {
    __shared__ float sh[256];
    __shared__ float red[8];
    int row = blockIdx.x, t = threadIdx.x;
    int lane = t & 31, wid = t >> 5;

    float v = s[row * 256 + t];

    // block max
    float m = v;
    #pragma unroll
    for (int off = 16; off > 0; off >>= 1) m = fmaxf(m, __shfl_xor(m, off, 32));
    if (lane == 0) red[wid] = m;
    __syncthreads();
    if (t == 0) {
        float M = red[0];
        #pragma unroll
        for (int i = 1; i < 8; ++i) M = fmaxf(M, red[i]);
        red[0] = M;
    }
    __syncthreads();
    float M = red[0];
    __syncthreads();

    // block sum of exp
    float p = __expf(v - M);
    float sum = p;
    #pragma unroll
    for (int off = 16; off > 0; off >>= 1) sum += __shfl_xor(sum, off, 32);
    if (lane == 0) red[wid] = sum;
    __syncthreads();
    if (t == 0) {
        float S = 0.0f;
        #pragma unroll
        for (int i = 0; i < 8; ++i) S += red[i];
        red[0] = S;
    }
    __syncthreads();
    float S = red[0];

    sh[t] = p / S;
    __syncthreads();

    // xout[row*256 + j][t] += sk[j], coalesced across t
    float* base = xout + (size_t)row * 256 * OUT_C + t;
    for (int j = 0; j < 256; ++j)
        base[(size_t)j * OUT_C] += sh[j];
}

// ============================================================
extern "C" void kernel_launch(void* const* d_in, const int* in_sizes, int n_in,
                              void* d_out, int out_size, void* d_ws, size_t ws_size,
                              hipStream_t stream) {
    const float* x        = (const float*)d_in[0];
    const int*   ei       = (const int*)d_in[2];
    const float* W_self   = (const float*)d_in[5];
    const float* b_self   = (const float*)d_in[6];
    const float* W_l      = (const float*)d_in[7];
    const float* b_l      = (const float*)d_in[8];
    const float* W_r      = (const float*)d_in[9];
    const float* b_r      = (const float*)d_in[10];
    const float* att      = (const float*)d_in[11];
    const float* conv_b   = (const float*)d_in[12];
    const float* W_cls    = (const float*)d_in[13];
    const float* b_cls    = (const float*)d_in[14];
    float* xout = (float*)d_out;

    // workspace carve-up
    char* ws = (char*)d_ws;
    size_t off = 0;
    auto alloc = [&](size_t bytes) { void* p = ws + off; off = (off + bytes + 255) & ~(size_t)255; return p; };
    __bf16*   xb    = (__bf16*)  alloc((size_t)N_NODES * IN_C * 2);      // 64 MB
    __bf16*   wpack = (__bf16*)  alloc((size_t)3 * 256 * 32 * 16 * 2);   // 768 KB
    float*    xl    = (float*)   alloc((size_t)N_NODES * OUT_C * 4);     // 64 MB
    float*    xr    = (float*)   alloc((size_t)N_NODES * OUT_C * 4);     // 64 MB
    float*    elog  = (float*)   alloc((size_t)E_TOT * 4);
    unsigned* mkey  = (unsigned*)alloc((size_t)N_NODES * 4);
    float*    denom = (float*)   alloc((size_t)N_NODES * 4);
    float*    sbuf  = (float*)   alloc((size_t)N_NODES * 4);

    // 1) x -> bf16
    cvt_x_kernel<<<(N_NODES * IN_C / 4 + 255) / 256, 256, 0, stream>>>(x, xb, N_NODES * IN_C);
    // 2) pack weights into WMMA B layout
    pack_w_kernel<<<(3 * 256 * 32 + 255) / 256, 256, 0, stream>>>(W_l, W_r, W_self, wpack);
    // 3) zero softmax state
    zero_kernel<<<N_NODES / 256, 256, 0, stream>>>(mkey, denom);
    // 4) fused triple GEMM (WMMA, LDS-staged B, two-phase pipeline)
    gemm3_kernel<<<2048, 256, 0, stream>>>(xb, wpack, b_l, b_r, b_self, conv_b, xl, xr, xout);
    // 5) edge logits + segment max
    edge_logit_kernel<<<(E_TOT * 32) / 256, 256, 0, stream>>>(xl, xr, att, ei, elog, mkey);
    // 6) softmax denominators
    edge_denom_kernel<<<(E_TOT + 255) / 256, 256, 0, stream>>>(elog, mkey, denom, ei);
    // 7) weighted neighbor aggregation
    edge_agg_kernel<<<(E_TOT * 32) / 256, 256, 0, stream>>>(elog, mkey, denom, xl, ei, xout);
    // 8) classifier dot
    cls_dot_kernel<<<(N_NODES * 32) / 256, 256, 0, stream>>>(xout, W_cls, b_cls, sbuf);
    // 9) row softmax + broadcast add
    sk_softmax_kernel<<<256, 256, 0, stream>>>(sbuf, xout);
}
